// MSARowAttentionWithPairBias_52183852646488
// MI455X (gfx1250) — matrix operand order repaired and checked by reading, not compiled, not run on hardware
//
#include <hip/hip_runtime.h>
#include <hip/hip_bf16.h>
#include <math.h>

typedef __attribute__((ext_vector_type(16))) __bf16 bf16x16;
typedef __attribute__((ext_vector_type(8)))  __bf16 bf16x8;
typedef __attribute__((ext_vector_type(8)))  float  f32x8;

static constexpr int S = 128, I = 256, C = 32, H = 8;
static constexpr int ROWS = S * I;           // 32768
static constexpr int HC   = H * C;           // 256
static constexpr float SCALE = 0.17677669529663687f; // 32^-0.5

#define DEVINL __device__ __forceinline__

// A fragment 16x32 bf16 from row-major [M][K] source (rowStride elements).
// lanes 0-15: row=lane, K chunks {0-7,16-23}; lanes 16-31: row=lane-16, {8-15,24-31}
DEVINL bf16x16 loadA(const __bf16* base, int rowStride, int lane) {
  int m  = lane & 15;
  int kb = (lane >> 4) << 3;                 // 0 or 8
  const __bf16* r = base + m * rowStride + kb;
  bf16x8 lo = *(const bf16x8*)(r);
  bf16x8 hi = *(const bf16x8*)(r + 16);
  bf16x16 a;
#pragma unroll
  for (int t = 0; t < 8; ++t) { a[t] = lo[t]; a[t + 8] = hi[t]; }
  return a;
}

// B fragment 32x16 bf16 from K-transposed [N][K] source (rowStride elements).
// lanes 0-15: col n=lane, K chunks {0-7,16-23}; lanes 16-31: col n=lane-16, {8-15,24-31}
DEVINL bf16x16 loadB(const __bf16* base, int rowStride, int lane) {
  int n  = lane & 15;
  int kr = (lane >> 4) << 3;                 // 0 or 8
  const __bf16* r = base + n * rowStride + kr;
  bf16x8 lo = *(const bf16x8*)(r);
  bf16x8 hi = *(const bf16x8*)(r + 16);
  bf16x16 b;
#pragma unroll
  for (int t = 0; t < 8; ++t) { b[t] = lo[t]; b[t + 8] = hi[t]; }
  return b;
}

DEVINL f32x8 wmma_bf16(bf16x16 a, bf16x16 b, f32x8 c) {
  return __builtin_amdgcn_wmma_f32_16x16x32_bf16(false, a, false, b,
                                                 (short)0, c, false, false);
}

// ---------------- Kernel 1: weight transpose + bf16 convert ----------------
__global__ void k_prep(const float* __restrict__ Wq, const float* __restrict__ Wk,
                       const float* __restrict__ Wv, const float* __restrict__ Wg,
                       const float* __restrict__ Wo,
                       __bf16* WqT, __bf16* WkT, __bf16* WvT, __bf16* WgT,
                       __bf16* WoT) {
  int tid = threadIdx.x;
  for (int idx = tid; idx < C * HC; idx += blockDim.x) {   // (32,256) -> [n][c]
    int n = idx >> 5, c = idx & 31;
    WqT[n * C + c] = (__bf16)Wq[c * HC + n];
    WkT[n * C + c] = (__bf16)Wk[c * HC + n];
    WvT[n * C + c] = (__bf16)Wv[c * HC + n];
    WgT[n * C + c] = (__bf16)Wg[c * HC + n];
  }
  for (int idx = tid; idx < HC * C; idx += blockDim.x) {   // (256,32) -> [n][k]
    int n = idx >> 8, k = idx & 255;
    WoT[n * HC + k] = (__bf16)Wo[k * C + n];
  }
}

// -------------- Kernel 2: LayerNorm(m) + Q/K/V/Gate projections ------------
__global__ __launch_bounds__(256)
void k_ln_proj(const float* __restrict__ m, const float* __restrict__ g_m,
               const float* __restrict__ b_m,
               const __bf16* __restrict__ WqT, const __bf16* __restrict__ WkT,
               const __bf16* __restrict__ WvT, const __bf16* __restrict__ WgT,
               __bf16* qb, __bf16* kb, __bf16* vtb, __bf16* gateb) {
  __shared__ __align__(16) __bf16 mnT[128 * C];    // 8 KB
  __shared__ __align__(16) __bf16 wT[4][HC * C];   // 64 KB
  int tid = threadIdx.x;
  int r0  = blockIdx.x * 128;

  // cooperative LDS staging of the 4 transposed weights (b128 copies)
  {
    const uint4* s0 = (const uint4*)WqT; const uint4* s1 = (const uint4*)WkT;
    const uint4* s2 = (const uint4*)WvT; const uint4* s3 = (const uint4*)WgT;
    uint4* d0 = (uint4*)wT[0]; uint4* d1 = (uint4*)wT[1];
    uint4* d2 = (uint4*)wT[2]; uint4* d3 = (uint4*)wT[3];
    for (int idx = tid; idx < (HC * C) / 8; idx += 256) {
      d0[idx] = s0[idx]; d1[idx] = s1[idx]; d2[idx] = s2[idx]; d3[idx] = s3[idx];
    }
  }
  if (tid < 128) {                                 // LayerNorm, one row/thread
    const float* row = m + (size_t)(r0 + tid) * C;
    float x[C]; float mu = 0.f;
#pragma unroll
    for (int c = 0; c < C; ++c) { x[c] = row[c]; mu += x[c]; }
    mu *= (1.f / C);
    float var = 0.f;
#pragma unroll
    for (int c = 0; c < C; ++c) { float d = x[c] - mu; var += d * d; }
    float rstd = rsqrtf(var * (1.f / C) + 1e-5f);
#pragma unroll
    for (int c = 0; c < C; ++c)
      mnT[tid * C + c] = (__bf16)((x[c] - mu) * rstd * g_m[c] + b_m[c]);
  }
  __syncthreads();

  int lane = tid & 31, wave = tid >> 5;
  int rt = wave * 16;
  bf16x16 afrag = loadA(mnT + rt * C, C, lane);

  for (int nt = 0; nt < 16; ++nt) {
    bf16x16 bq = loadB(&wT[0][nt * 16 * C], C, lane);
    bf16x16 bk = loadB(&wT[1][nt * 16 * C], C, lane);
    bf16x16 bv = loadB(&wT[2][nt * 16 * C], C, lane);
    bf16x16 bg = loadB(&wT[3][nt * 16 * C], C, lane);
    f32x8 z = {};
    f32x8 aq = wmma_bf16(afrag, bq, z);
    f32x8 ak = wmma_bf16(afrag, bk, z);
    f32x8 av = wmma_bf16(afrag, bv, z);
    f32x8 ag = wmma_bf16(afrag, bg, z);
#pragma unroll
    for (int r = 0; r < 8; ++r) {
      int mrow = ((lane >> 4) << 3) + r;
      int R = r0 + rt + mrow;
      int s = R >> 8, i = R & 255;                 // I == 256
      int n = nt * 16 + (lane & 15);
      int h = n >> 5, c = n & 31;
      size_t ihc = (size_t)((s * H + h) * I + i) * C + c;
      qb[ihc] = (__bf16)aq[r];
      kb[ihc] = (__bf16)ak[r];
      vtb[(size_t)((s * H + h) * C + c) * I + i] = (__bf16)av[r];
      float e = __expf(-ag[r]);
      gateb[ihc] = (__bf16)__builtin_amdgcn_rcpf(1.f + e);
    }
  }
}

// -------------- Kernel 3: LayerNorm(z) + pair bias (VALU, tiny) ------------
__global__ __launch_bounds__(256)
void k_bias(const float* __restrict__ z, const float* __restrict__ g_z,
            const float* __restrict__ b_z, const float* __restrict__ Wb,
            float* bias) {
  int r = blockIdx.x * blockDim.x + threadIdx.x;          // r = i*I + j
  if (r >= I * I) return;
  const float* row = z + (size_t)r * C;
  float x[C]; float mu = 0.f;
#pragma unroll
  for (int c = 0; c < C; ++c) { x[c] = row[c]; mu += x[c]; }
  mu *= (1.f / C);
  float var = 0.f;
#pragma unroll
  for (int c = 0; c < C; ++c) { float d = x[c] - mu; var += d * d; }
  float rstd = rsqrtf(var * (1.f / C) + 1e-5f);
  float acc[H] = {};
#pragma unroll
  for (int c = 0; c < C; ++c) {
    float zn = (x[c] - mu) * rstd * g_z[c] + b_z[c];
#pragma unroll
    for (int h = 0; h < H; ++h) acc[h] += zn * Wb[c * H + h];
  }
#pragma unroll
  for (int h = 0; h < H; ++h) bias[(size_t)h * I * I + r] = acc[h];
}

// -------------- Kernel 4: per-(s,h) attention, fully WMMA ------------------
__global__ __launch_bounds__(256)
void k_attn(const __bf16* __restrict__ qb, const __bf16* __restrict__ kb,
            const __bf16* __restrict__ vtb, const __bf16* __restrict__ gateb,
            const float* __restrict__ bias, __bf16* ob) {
  __shared__ __align__(16) __bf16 qs[I * C];        // 16 KB
  __shared__ __align__(16) __bf16 ks[I * C];        // 16 KB
  __shared__ __align__(16) __bf16 vts[C * I];       // 16 KB
  __shared__ __align__(16) __bf16 probs[8][16 * I]; // 64 KB

  int tid = threadIdx.x, lane = tid & 31, wave = tid >> 5;
  int s = blockIdx.x >> 3, h = blockIdx.x & 7;
  size_t base = (size_t)(s * H + h) * I * C;

  {
    const uint4* q4 = (const uint4*)(qb + base);
    const uint4* k4 = (const uint4*)(kb + base);
    const uint4* v4 = (const uint4*)(vtb + base);
    uint4* qd = (uint4*)qs; uint4* kd = (uint4*)ks; uint4* vd = (uint4*)vts;
    for (int idx = tid; idx < (I * C) / 8; idx += 256) {
      qd[idx] = q4[idx]; kd[idx] = k4[idx]; vd[idx] = v4[idx];
    }
  }
  __syncthreads();

  const float* biasH = bias + (size_t)h * I * I;
  __bf16* pw = probs[wave];

  for (int half = 0; half < 2; ++half) {
    int i0 = (wave * 2 + half) * 16;
    __builtin_prefetch(biasH + (size_t)i0 * I, 0, 1);
    bf16x16 aq = loadA(qs + i0 * C, C, lane);

    // scores: 16 j-tiles, K=C=32 in a single WMMA each
    f32x8 sc[16];
#pragma unroll
    for (int jt = 0; jt < 16; ++jt) {
      bf16x16 bk = loadB(ks + jt * 16 * C, C, lane);
      f32x8 zc = {};
      sc[jt] = wmma_bf16(aq, bk, zc);
    }
    // scale + pair bias
#pragma unroll
    for (int jt = 0; jt < 16; ++jt) {
#pragma unroll
      for (int r = 0; r < 8; ++r) {
        int mrow = ((lane >> 4) << 3) + r;
        int i = i0 + mrow;
        int j = jt * 16 + (lane & 15);
        sc[jt][r] = sc[jt][r] * SCALE + biasH[(size_t)i * I + j];
      }
    }
    // softmax over j (16 tiles in regs + 16-lane shuffle reduce)
    float mx[8], sum[8];
#pragma unroll
    for (int r = 0; r < 8; ++r) {
      float v = sc[0][r];
#pragma unroll
      for (int jt = 1; jt < 16; ++jt) v = fmaxf(v, sc[jt][r]);
#pragma unroll
      for (int msk = 1; msk < 16; msk <<= 1) v = fmaxf(v, __shfl_xor(v, msk, 32));
      mx[r] = v; sum[r] = 0.f;
    }
#pragma unroll
    for (int jt = 0; jt < 16; ++jt)
#pragma unroll
      for (int r = 0; r < 8; ++r) {
        float e = __expf(sc[jt][r] - mx[r]);
        sc[jt][r] = e; sum[r] += e;
      }
#pragma unroll
    for (int r = 0; r < 8; ++r) {
#pragma unroll
      for (int msk = 1; msk < 16; msk <<= 1) sum[r] += __shfl_xor(sum[r], msk, 32);
      sum[r] = __builtin_amdgcn_rcpf(sum[r]);
    }
    // bf16 probs -> per-wave LDS tile (row-major 16 x 256) for A-fragments
#pragma unroll
    for (int jt = 0; jt < 16; ++jt)
#pragma unroll
      for (int r = 0; r < 8; ++r) {
        int mrow = ((lane >> 4) << 3) + r;
        int j = jt * 16 + (lane & 15);
        pw[mrow * I + j] = (__bf16)(sc[jt][r] * sum[r]);
      }
    asm volatile("s_wait_dscnt 0" ::: "memory");

    // O = P @ V (16 x 32), gate, store
#pragma unroll
    for (int ct = 0; ct < 2; ++ct) {
      f32x8 oacc = {};
#pragma unroll
      for (int kc = 0; kc < 8; ++kc) {
        bf16x16 ap = loadA(pw + kc * 32, I, lane);
        bf16x16 bv = loadB(vts + ct * 16 * I + kc * 32, I, lane);
        oacc = wmma_bf16(ap, bv, oacc);
      }
#pragma unroll
      for (int r = 0; r < 8; ++r) {
        int mrow = ((lane >> 4) << 3) + r;
        int i = i0 + mrow;
        int c = ct * 16 + (lane & 15);
        float g = (float)gateb[base + (size_t)i * C + c];
        ob[(size_t)(s * I + i) * HC + h * C + c] = (__bf16)(oacc[r] * g);
      }
    }
  }
}

// -------------- Kernel 5: output projection + bias -------------------------
__global__ __launch_bounds__(256)
void k_outproj(const __bf16* __restrict__ ob, const __bf16* __restrict__ WoT,
               const float* __restrict__ bo, float* out) {
  __shared__ __align__(16) __bf16 wo[C * HC];       // 16 KB
  int tid = threadIdx.x, lane = tid & 31, wave = tid >> 5;
  int R0 = blockIdx.x * 128 + wave * 16;
  {
    const uint4* src = (const uint4*)WoT;
    uint4* dst = (uint4*)wo;
    for (int idx = tid; idx < (C * HC) / 8; idx += 256) dst[idx] = src[idx];
  }
  __syncthreads();

#pragma unroll
  for (int ct = 0; ct < 2; ++ct) {
    f32x8 acc = {};
#pragma unroll
    for (int kc = 0; kc < 8; ++kc) {
      bf16x16 a = loadA(ob + (size_t)R0 * HC + kc * 32, HC, lane);
      bf16x16 b = loadB(wo + ct * 16 * HC + kc * 32, HC, lane);
      acc = wmma_bf16(a, b, acc);
    }
#pragma unroll
    for (int r = 0; r < 8; ++r) {
      int mrow = ((lane >> 4) << 3) + r;
      int R = R0 + mrow;
      int c = ct * 16 + (lane & 15);
      out[(size_t)R * C + c] = acc[r] + bo[c];
    }
  }
}

extern "C" void kernel_launch(void* const* d_in, const int* in_sizes, int n_in,
                              void* d_out, int out_size, void* d_ws, size_t ws_size,
                              hipStream_t stream) {
  const float* m   = (const float*)d_in[0];
  const float* z   = (const float*)d_in[1];
  const float* g_m = (const float*)d_in[2];
  const float* b_m = (const float*)d_in[3];
  const float* g_z = (const float*)d_in[4];
  const float* b_z = (const float*)d_in[5];
  const float* Wq  = (const float*)d_in[6];
  const float* Wk  = (const float*)d_in[7];
  const float* Wv  = (const float*)d_in[8];
  const float* Wb  = (const float*)d_in[9];
  const float* Wg  = (const float*)d_in[10];
  const float* Wo  = (const float*)d_in[11];
  const float* bo  = (const float*)d_in[12];
  float* out = (float*)d_out;

  char* ws = (char*)d_ws;
  const size_t nQ = (size_t)S * H * I * C;                 // 8,388,608 elems
  __bf16* qb    = (__bf16*)(ws);
  __bf16* kb    = (__bf16*)(ws + nQ * 2);
  __bf16* vtb   = (__bf16*)(ws + nQ * 4);
  __bf16* gateb = (__bf16*)(ws + nQ * 6);
  __bf16* ob    = (__bf16*)(ws + nQ * 8);
  float*  bias  = (float*)(ws + nQ * 10);
  __bf16* WqT   = (__bf16*)(ws + nQ * 10 + (size_t)H * I * I * 4);
  __bf16* WkT = WqT + C * HC;
  __bf16* WvT = WkT + C * HC;
  __bf16* WgT = WvT + C * HC;
  __bf16* WoT = WgT + C * HC;

  hipLaunchKernelGGL(k_prep, dim3(1), dim3(256), 0, stream,
                     Wq, Wk, Wv, Wg, Wo, WqT, WkT, WvT, WgT, WoT);
  hipLaunchKernelGGL(k_ln_proj, dim3(ROWS / 128), dim3(256), 0, stream,
                     m, g_m, b_m, WqT, WkT, WvT, WgT, qb, kb, vtb, gateb);
  hipLaunchKernelGGL(k_bias, dim3((I * I) / 256), dim3(256), 0, stream,
                     z, g_z, b_z, Wb, bias);
  hipLaunchKernelGGL(k_attn, dim3(S * H), dim3(256), 0, stream,
                     qb, kb, vtb, gateb, bias, ob);
  hipLaunchKernelGGL(k_outproj, dim3(ROWS / 128), dim3(256), 0, stream,
                     ob, WoT, bo, out);
}
